// MultiHeadAttention_5686536700414
// MI455X (gfx1250) — matrix operand order repaired and checked by reading, not compiled, not run on hardware
//
#include <hip/hip_runtime.h>
#include <hip/hip_bf16.h>

// ---------------------------------------------------------------------------
// Types / helpers
// ---------------------------------------------------------------------------
typedef __bf16 bf16;
typedef __bf16 bf16x16 __attribute__((ext_vector_type(16)));
typedef __bf16 bf16x8  __attribute__((ext_vector_type(8)));
typedef float  fx8     __attribute__((ext_vector_type(8)));
typedef float  fx4     __attribute__((ext_vector_type(4)));

#define WMMA_BF16(a, b, c)                                                     \
  __builtin_amdgcn_wmma_f32_16x16x32_bf16(false, (a), false, (b), (short)0,    \
                                          (c), false, false)

// Row-max across the 16 lanes of a wave-half (xor masks 1,2,4,8 stay inside
// each half of the wave32).
static __device__ __forceinline__ float half16_max(float v) {
  v = fmaxf(v, __shfl_xor(v, 1, 32));
  v = fmaxf(v, __shfl_xor(v, 2, 32));
  v = fmaxf(v, __shfl_xor(v, 4, 32));
  v = fmaxf(v, __shfl_xor(v, 8, 32));
  return v;
}

// ---------------------------------------------------------------------------
// Elementwise fp32 -> bf16 (8 elements / thread, vectorized loads/stores)
// ---------------------------------------------------------------------------
__global__ __launch_bounds__(256) void cvt_f32_bf16(const float* __restrict__ in,
                                                    bf16* __restrict__ out,
                                                    int n) {
  const int i = (blockIdx.x * blockDim.x + threadIdx.x) * 8;
  if (i + 7 < n) {
    fx4 a = *(const fx4*)(in + i);
    fx4 b = *(const fx4*)(in + i + 4);
    bf16x8 o;
#pragma unroll
    for (int j = 0; j < 4; ++j) { o[j] = (bf16)a[j]; o[4 + j] = (bf16)b[j]; }
    *(bf16x8*)(out + i) = o;
  }
}

// ---------------------------------------------------------------------------
// GEMM:  O[M,N] = X[M,K] @ W[N,K]^T, all-bf16 operands, f32 accumulation.
// One wave computes a 16x64 strip (A fragment reused 4x). 8 waves / block
// cover N = 512. TRANS_OUT stores O transposed as [(b*N + n), s] for V.
// ---------------------------------------------------------------------------
template <typename OutT, bool TRANS_OUT>
__global__ __launch_bounds__(256) void gemm_bf16(const bf16* __restrict__ X,
                                                 const bf16* __restrict__ W,
                                                 OutT* __restrict__ O, int M,
                                                 int N, int K, int Sdim) {
  const int lane = threadIdx.x & 31;
  const int wave = threadIdx.x >> 5;
  const int hl   = lane >> 4;
  const int l15  = lane & 15;
  const int tileM = blockIdx.x * 16;
  const int tileN = wave * 64;

  const bf16* xrow = X + (size_t)(tileM + l15) * K;  // A: lane = row
  const bf16* wrow = W + (size_t)(tileN + l15) * K;  // B: lane = col (W is NxK)

  fx8 acc0 = {}, acc1 = {}, acc2 = {}, acc3 = {};
#pragma unroll 2
  for (int kb = 0; kb < K; kb += 32) {
    // A fragment (16x32): 16 contiguous k at 16*hl (two 8-elt chunks)
    const int ka = kb + hl * 16;
    bf16x8 alo = *(const bf16x8*)(xrow + ka);
    bf16x8 ahi = *(const bf16x8*)(xrow + ka + 8);
    bf16x16 va;
#pragma unroll
    for (int i = 0; i < 8; ++i) { va[i] = alo[i]; va[8 + i] = ahi[i]; }
    // Four B fragments (32x16): 16 contiguous k at 16*hl, cols tileN+j*16+l15
    const int kn = kb + hl * 16;
    bf16x16 vb0 = *(const bf16x16*)(wrow + kn);
    bf16x16 vb1 = *(const bf16x16*)(wrow + (size_t)16 * K + kn);
    bf16x16 vb2 = *(const bf16x16*)(wrow + (size_t)32 * K + kn);
    bf16x16 vb3 = *(const bf16x16*)(wrow + (size_t)48 * K + kn);
    acc0 = WMMA_BF16(va, vb0, acc0);
    acc1 = WMMA_BF16(va, vb1, acc1);
    acc2 = WMMA_BF16(va, vb2, acc2);
    acc3 = WMMA_BF16(va, vb3, acc3);
  }

  if (TRANS_OUT) {
    // O[(b*N + n)*S + s]; each lane's 8 rows are contiguous in s.
    const int mbase = tileM + hl * 8;
    const int bidx  = mbase / Sdim;
    const int sbase = mbase % Sdim;
    const fx8 accs[4] = {acc0, acc1, acc2, acc3};
#pragma unroll
    for (int j = 0; j < 4; ++j) {
      bf16x8 pack;
#pragma unroll
      for (int r = 0; r < 8; ++r) pack[r] = (bf16)accs[j][r];
      bf16* optr = (bf16*)O +
                   ((size_t)bidx * N + tileN + j * 16 + l15) * Sdim + sbase;
      *(bf16x8*)optr = pack;
    }
  } else {
    OutT* orow = O + (size_t)(tileM + hl * 8) * N + tileN + l15;
#pragma unroll
    for (int r = 0; r < 8; ++r) {
      orow[(size_t)r * N]      = (OutT)acc0[r];
      orow[(size_t)r * N + 16] = (OutT)acc1[r];
      orow[(size_t)r * N + 32] = (OutT)acc2[r];
      orow[(size_t)r * N + 48] = (OutT)acc3[r];
    }
  }
}

// ---------------------------------------------------------------------------
// Flash attention. Qp/Kp bf16 [B,S,H*DK]; Vt bf16 transposed [B, H*DK, S].
// One wave owns 16 query rows of one (b,h); streams keys in blocks of 64 with
// online softmax. Row-sums computed via P x ones WMMA (C-layout puts row r's
// sum at accumulator element r). Mask (int32 [B,H,S,S]) read exactly once.
// ---------------------------------------------------------------------------
__global__ __launch_bounds__(256) void flash_attn(
    const bf16* __restrict__ Qp, const bf16* __restrict__ Kp,
    const bf16* __restrict__ Vt, const int* __restrict__ mask,
    bf16* __restrict__ AO) {
  constexpr int S = 2048, H = 8, DM = 512, DK = 64;
  const int lane = threadIdx.x & 31;
  const int wave = threadIdx.x >> 5;
  const int hl   = lane >> 4;
  const int l15  = lane & 15;

  const int gid = blockIdx.x * 8 + wave;  // [0, B*H*S/16)
  const int q0  = (gid & 127) * 16;
  const int bh  = gid >> 7;
  const int b   = bh >> 3;
  const int h   = bh & 7;

  const float scale   = 0.04419417382415922f;  // 1/sqrt(d_model=512)
  const float NEG_INF = -__builtin_inff();

  // Per-wave LDS staging of P (16x64 bf16): C-fragment -> A-fragment relayout.
  __shared__ bf16 plds[8][16 * 64];
  bf16* pw = plds[wave];

  // Q A-fragments, pre-scaled by 1/sqrt(d_model) (folds the score scaling
  // into the operand so the softmax path needs no per-element multiply).
  const bf16* qrow = Qp + ((size_t)b * S + q0 + l15) * DM + h * DK;
  bf16x16 qa0, qa1;
  {
    const int k0 = hl * 16;
    bf16x8 lo = *(const bf16x8*)(qrow + k0);
    bf16x8 hi = *(const bf16x8*)(qrow + k0 + 8);
#pragma unroll
    for (int i = 0; i < 8; ++i) {
      qa0[i]     = (bf16)((float)lo[i] * scale);
      qa0[8 + i] = (bf16)((float)hi[i] * scale);
    }
    lo = *(const bf16x8*)(qrow + 32 + k0);
    hi = *(const bf16x8*)(qrow + 32 + k0 + 8);
#pragma unroll
    for (int i = 0; i < 8; ++i) {
      qa1[i]     = (bf16)((float)lo[i] * scale);
      qa1[8 + i] = (bf16)((float)hi[i] * scale);
    }
  }

  // All-ones B fragment for row-sum WMMAs.
  bf16x16 ones;
#pragma unroll
  for (int i = 0; i < 16; ++i) ones[i] = (bf16)1.0f;

  float rm[8];
#pragma unroll
  for (int r = 0; r < 8; ++r) rm[r] = NEG_INF;
  fx8 rlacc = {};   // running softmax denominators (one per owned row)
  fx8 oac[4] = {};  // 16x64 output accumulator

  const size_t kbase = (size_t)b * S * DM + h * DK;         // Kp row-major
  const bf16*  vtb   = Vt + ((size_t)b * DM + h * DK) * S;  // Vt: [d][s]
  const int*   mrow  = mask + ((size_t)(b * H + h) * S + q0) * S;

  for (int kb = 0; kb < S; kb += 64) {
    // -------- scores: four 16x16 tiles (keys kb .. kb+63)
    fx8 s0, s1, s2, s3;
    {
      const bf16* k0p = Kp + kbase + (size_t)(kb + l15) * DM;
      const bf16* k1p = k0p + (size_t)16 * DM;
      const bf16* k2p = k0p + (size_t)32 * DM;
      const bf16* k3p = k0p + (size_t)48 * DM;
      const int d0 = hl * 16;
      s0 = WMMA_BF16(qa0, *(const bf16x16*)(k0p + d0), (fx8){});
      s0 = WMMA_BF16(qa1, *(const bf16x16*)(k0p + 32 + d0), s0);
      s1 = WMMA_BF16(qa0, *(const bf16x16*)(k1p + d0), (fx8){});
      s1 = WMMA_BF16(qa1, *(const bf16x16*)(k1p + 32 + d0), s1);
      s2 = WMMA_BF16(qa0, *(const bf16x16*)(k2p + d0), (fx8){});
      s2 = WMMA_BF16(qa1, *(const bf16x16*)(k2p + 32 + d0), s2);
      s3 = WMMA_BF16(qa0, *(const bf16x16*)(k3p + d0), (fx8){});
      s3 = WMMA_BF16(qa1, *(const bf16x16*)(k3p + 32 + d0), s3);
    }

    if (kb + 64 < S) {  // gfx1250 global_prefetch_b8 for next K/V tiles
      __builtin_prefetch(Kp + kbase + (size_t)(kb + 64 + l15) * DM, 0, 1);
      __builtin_prefetch(vtb + (size_t)l15 * S + kb + 64, 0, 1);
    }

    // -------- mask + online softmax update (per owned row)
#pragma unroll
    for (int r = 0; r < 8; ++r) {
      const int* mr = mrow + (size_t)(r + 8 * hl) * S + kb;
      float v0 = mr[l15]      ? s0[r] : NEG_INF;
      float v1 = mr[16 + l15] ? s1[r] : NEG_INF;
      float v2 = mr[32 + l15] ? s2[r] : NEG_INF;
      float v3 = mr[48 + l15] ? s3[r] : NEG_INF;
      float tmax = half16_max(fmaxf(fmaxf(v0, v1), fmaxf(v2, v3)));
      float mnew = fmaxf(rm[r], tmax);
      float f  = (rm[r] == NEG_INF) ? 0.f : __expf(rm[r] - mnew);
      float p0 = (mnew == NEG_INF) ? 0.f : __expf(v0 - mnew);
      float p1 = (mnew == NEG_INF) ? 0.f : __expf(v1 - mnew);
      float p2 = (mnew == NEG_INF) ? 0.f : __expf(v2 - mnew);
      float p3 = (mnew == NEG_INF) ? 0.f : __expf(v3 - mnew);
      rm[r] = mnew;
      rlacc[r] *= f;
      oac[0][r] *= f; oac[1][r] *= f; oac[2][r] *= f; oac[3][r] *= f;
      const int row = r + 8 * hl;
      pw[row * 64 + l15]      = (bf16)p0;
      pw[row * 64 + 16 + l15] = (bf16)p1;
      pw[row * 64 + 32 + l15] = (bf16)p2;
      pw[row * 64 + 48 + l15] = (bf16)p3;
    }

    // -------- C-layout P -> A-layout P via LDS round trip (per-wave region)
    __builtin_amdgcn_wave_barrier();
    __syncthreads();
    bf16x16 pa0, pa1;
    {
      const bf16* pr = pw + l15 * 64 + hl * 16;
      bf16x8 lo = *(const bf16x8*)(pr);
      bf16x8 hi = *(const bf16x8*)(pr + 8);
#pragma unroll
      for (int i = 0; i < 8; ++i) { pa0[i] = lo[i]; pa0[8 + i] = hi[i]; }
      lo = *(const bf16x8*)(pr + 32);
      hi = *(const bf16x8*)(pr + 40);
#pragma unroll
      for (int i = 0; i < 8; ++i) { pa1[i] = lo[i]; pa1[8 + i] = hi[i]; }
    }
    __syncthreads();

    // -------- denominators: rowsum(P) via P x ones (replaces lane shuffles)
    rlacc = WMMA_BF16(pa0, ones, rlacc);
    rlacc = WMMA_BF16(pa1, ones, rlacc);

    // -------- O += P(16x64) x V(64x64): B frags are contiguous rows of Vt
#pragma unroll
    for (int j = 0; j < 4; ++j) {
      const bf16* vr = vtb + (size_t)(j * 16 + l15) * S + kb + hl * 16;
      oac[j] = WMMA_BF16(pa0, *(const bf16x16*)(vr), oac[j]);
      oac[j] = WMMA_BF16(pa1, *(const bf16x16*)(vr + 32), oac[j]);
    }
  }

  // -------- normalize + write attention output bf16 [B, S, H*DK]
#pragma unroll
  for (int r = 0; r < 8; ++r) {
    const float inv = rlacc[r] > 0.f ? 1.0f / rlacc[r] : 0.f;
    bf16* orow = AO + ((size_t)b * S + q0 + r + 8 * hl) * DM + h * DK + l15;
    orow[0]  = (bf16)(oac[0][r] * inv);
    orow[16] = (bf16)(oac[1][r] * inv);
    orow[32] = (bf16)(oac[2][r] * inv);
    orow[48] = (bf16)(oac[3][r] * inv);
  }
}

// ---------------------------------------------------------------------------
// Launch pipeline
// ---------------------------------------------------------------------------
extern "C" void kernel_launch(void* const* d_in, const int* in_sizes, int n_in,
                              void* d_out, int out_size, void* d_ws,
                              size_t ws_size, hipStream_t stream) {
  (void)in_sizes; (void)n_in; (void)out_size; (void)ws_size;
  const float* q    = (const float*)d_in[0];
  const float* k    = (const float*)d_in[1];
  const float* v    = (const float*)d_in[2];
  const int*   mask = (const int*)d_in[3];
  const float* Wq   = (const float*)d_in[4];
  const float* Wk   = (const float*)d_in[5];
  const float* Wv   = (const float*)d_in[6];
  const float* Wo   = (const float*)d_in[7];

  constexpr int B = 2, S = 2048, DM = 512, H = 8;
  const int M  = B * S;   // 4096
  const int NX = M * DM;  // 2,097,152 elements per activation
  const int NW = DM * DM; // 262,144 elements per weight

  // Workspace carve: qb/kb/vb 3x4MB, weights 4x0.5MB, Qp/Kp 8MB, Vt 4MB,
  // AObf 4MB  => ~30 MB total (bf16).
  char* ws = (char*)d_ws;
  bf16* qb  = (bf16*)ws;
  bf16* kbuf= qb  + (size_t)NX;
  bf16* vb  = kbuf+ (size_t)NX;
  bf16* wqb = vb  + (size_t)NX;
  bf16* wkb = wqb + (size_t)NW;
  bf16* wvb = wkb + (size_t)NW;
  bf16* wob = wvb + (size_t)NW;
  bf16* Qp  = wob + (size_t)NW;
  bf16* Kp  = Qp  + (size_t)NX;
  bf16* Vt  = Kp  + (size_t)NX;
  bf16* AOb = Vt  + (size_t)NX;

  const dim3 blk(256);
  const int cvtg_x = (NX / 8 + 255) / 256;
  const int cvtg_w = (NW / 8 + 255) / 256;
  cvt_f32_bf16<<<cvtg_x, blk, 0, stream>>>(q, qb, NX);
  cvt_f32_bf16<<<cvtg_x, blk, 0, stream>>>(k, kbuf, NX);
  cvt_f32_bf16<<<cvtg_x, blk, 0, stream>>>(v, vb, NX);
  cvt_f32_bf16<<<cvtg_w, blk, 0, stream>>>(Wq, wqb, NW);
  cvt_f32_bf16<<<cvtg_w, blk, 0, stream>>>(Wk, wkb, NW);
  cvt_f32_bf16<<<cvtg_w, blk, 0, stream>>>(Wv, wvb, NW);
  cvt_f32_bf16<<<cvtg_w, blk, 0, stream>>>(Wo, wob, NW);

  const dim3 ggrid(M / 16);  // 256 blocks; 8 waves x 16x64 covers N=512
  gemm_bf16<bf16, false><<<ggrid, blk, 0, stream>>>(qb, wqb, Qp, M, DM, DM, S);
  gemm_bf16<bf16, false><<<ggrid, blk, 0, stream>>>(kbuf, wkb, Kp, M, DM, DM, S);
  gemm_bf16<bf16, true><<<ggrid, blk, 0, stream>>>(vb, wvb, Vt, M, DM, DM, S);

  const int qtiles = B * H * (S / 16);  // 2048 waves
  flash_attn<<<dim3(qtiles / 8), blk, 0, stream>>>(Qp, Kp, Vt, mask, AOb);

  gemm_bf16<float, false><<<ggrid, blk, 0, stream>>>(AOb, wob, (float*)d_out,
                                                     M, DM, DM, S);
}
